// TemporalAttention_79877801771266
// MI455X (gfx1250) — matrix operand order repaired and checked by reading, not compiled, not run on hardware
//
#include <hip/hip_runtime.h>

// TemporalAttention for MI455X (gfx1250, wave32, WMMA f16->f32).
// B=32, C=128, L=2048. Flash-attention style: never materialize the 537MB
// attention matrix. K/V tiles are staged into LDS by the Tensor Data Mover
// (tensor_load_to_lds) with hardware row padding, double-buffered and
// overlapped with WMMA compute. Workspace: 3 * B*L*C f16 = 48 MB.

#define BB 32
#define CCH 128
#define LLEN 2048

typedef __attribute__((ext_vector_type(16))) _Float16 v16h;
typedef __attribute__((ext_vector_type(8)))  _Float16 v8h;
typedef __attribute__((ext_vector_type(8)))  float    v8f;
typedef __attribute__((ext_vector_type(4)))  unsigned int u32x4;
typedef __attribute__((ext_vector_type(8)))  int i32x8;
typedef __attribute__((ext_vector_type(4)))  int i32x4;

union V16 { v16h v; v8h h[2]; };

__device__ __forceinline__ v8h cvt8(const float4 a, const float4 b) {
  v8h r;
  r[0] = (_Float16)a.x; r[1] = (_Float16)a.y; r[2] = (_Float16)a.z; r[3] = (_Float16)a.w;
  r[4] = (_Float16)b.x; r[5] = (_Float16)b.y; r[6] = (_Float16)b.z; r[7] = (_Float16)b.w;
  return r;
}

// ---------------------------------------------------------------------------
// TDM: 2D tile load Global -> LDS with per-row LDS padding.
// Descriptor layout per CDNA5 ISA ch.8 (D# groups 0/1; groups 2/3 zero = 2D).
//   tile_d0   : elements per row (data_size = 2 bytes)
//   tile_d1   : number of rows
//   stride_d0 : row stride in elements (tensor_dim0_stride)
//   pad_iv    : pad_interval code (pad after 2^(code+1) DWORDs)
//   pad_am    : pad_amount code (code+1 DWORDs of padding)
// Toolchain: 6-arg builtin (uint32x4, int32x8, int32x4, int32x4, int32x8, i32).
// ---------------------------------------------------------------------------
__device__ __forceinline__ void tdm_load_2d(const _Float16* gsrc, unsigned lds_byte_addr,
                                            unsigned tile_d0, unsigned tile_d1,
                                            unsigned stride_d0, unsigned pad_iv,
                                            unsigned pad_am) {
  const unsigned long long ga = (unsigned long long)(uintptr_t)gsrc;
  u32x4 g0;
  g0[0] = 1u;                                       // count=1, user mode
  g0[1] = lds_byte_addr;                            // D#.lds_addr
  g0[2] = (unsigned)(ga & 0xffffffffu);             // global_addr[31:0]
  g0[3] = (unsigned)((ga >> 32) & 0x01ffffffu)      // global_addr[56:32]
        | (2u << 30);                               // type = 2 ("image")

  i32x8 g1;
  g1[0] = (int)((1u << 16)                          // data_size = 1 -> 2 bytes
        | (1u << 20)                                // pad_enable
        | (pad_iv << 22) | (pad_am << 25));
  g1[1] = (int)((tile_d0 & 0xffffu) << 16);         // tensor_dim0[15:0] (abar=0)
  g1[2] = (int)(((tile_d0 >> 16) & 0xffffu)         // tensor_dim0[31:16]
        | ((tile_d1 & 0xffffu) << 16));             // tensor_dim1[15:0]
  g1[3] = (int)(((tile_d1 >> 16) & 0xffffu)         // tensor_dim1[31:16]
        | ((tile_d0 & 0xffffu) << 16));             // tile_dim0
  g1[4] = (int)(tile_d1 & 0xffffu);                 // tile_dim1 (tile_dim2 = 0)
  g1[5] = (int)stride_d0;                           // tensor_dim0_stride[31:0]
  g1[6] = 0;                                        // stride hi / dim1_stride lo
  g1[7] = 0;
  const i32x4 z4 = {0, 0, 0, 0};
  const i32x8 z8 = {0, 0, 0, 0, 0, 0, 0, 0};
  __builtin_amdgcn_tensor_load_to_lds(g0, g1, z4, z4, z8, 0);
}

// ---------------------------------------------------------------------------
// Kernel 1: q/k/v = W*x + b as f16, WMMA 16x16x32.
// Grid: B*(L/16) blocks, 256 threads (8 waves). Wave w -> out channels [16w,16w+16).
// q,k stored [b][l][c]; v stored [b][c][l].
// ---------------------------------------------------------------------------
__global__ __launch_bounds__(256) void qkv_proj_kernel(
    const float* __restrict__ x,
    const float* __restrict__ Wq, const float* __restrict__ bq,
    const float* __restrict__ Wk, const float* __restrict__ bk,
    const float* __restrict__ Wv, const float* __restrict__ bv,
    _Float16* __restrict__ qh, _Float16* __restrict__ kh, _Float16* __restrict__ vh)
{
  __shared__ _Float16 xs[16][136];   // x tile transposed: [l][c], padded rows

  const int blk = blockIdx.x;
  const int b   = blk / (LLEN / 16);
  const int l0  = (blk % (LLEN / 16)) * 16;
  const int t   = threadIdx.x;

  // Stage x[b, :, l0:l0+16] -> xs[l][c] as f16 (transpose during staging).
  if (t < CCH) {
    const float4* src = (const float4*)(x + ((size_t)b * CCH + t) * LLEN + l0);
    float4 f0 = src[0], f1 = src[1], f2 = src[2], f3 = src[3];
    v8h h0 = cvt8(f0, f1), h1 = cvt8(f2, f3);
#pragma unroll
    for (int i = 0; i < 8; ++i) { xs[i][t] = h0[i]; xs[8 + i][t] = h1[i]; }
  }
  __syncthreads();

  const int wave  = t >> 5;
  const int lane  = t & 31;
  const int row16 = lane & 15;
  const int hi    = lane >> 4;      // 0 for lanes 0-15, 1 for lanes 16-31
  const int ow    = wave * 16;      // output-channel tile base

  v8f aq = {}, ak = {}, av = {};
#pragma unroll
  for (int ccb = 0; ccb < 4; ++ccb) {        // K (channel) chunks of 32
    const int cbase = ccb * 32;
    // B fragment (K=c, N=l) from LDS: halves K {hi*8..hi*8+7, 16+hi*8..16+hi*8+7}.
    V16 bx;
    bx.h[0] = *(const v8h*)&xs[row16][cbase + hi * 8];
    bx.h[1] = *(const v8h*)&xs[row16][cbase + 16 + hi * 8];
    // A fragments: rows of W, halves K {+koff8..+koff8+7, +16+koff8..}.
    const size_t wrow = (size_t)(ow + row16) * CCH;
    const int k0 = cbase + hi * 8;
    const int k1 = cbase + 16 + hi * 8;
    V16 a;
    { const float4* p0 = (const float4*)(Wq + wrow + k0);
      const float4* p1 = (const float4*)(Wq + wrow + k1);
      a.h[0] = cvt8(p0[0], p0[1]); a.h[1] = cvt8(p1[0], p1[1]); }
    aq = __builtin_amdgcn_wmma_f32_16x16x32_f16(false, a.v, false, bx.v, (short)0, aq, false, false);
    { const float4* p0 = (const float4*)(Wk + wrow + k0);
      const float4* p1 = (const float4*)(Wk + wrow + k1);
      a.h[0] = cvt8(p0[0], p0[1]); a.h[1] = cvt8(p1[0], p1[1]); }
    ak = __builtin_amdgcn_wmma_f32_16x16x32_f16(false, a.v, false, bx.v, (short)0, ak, false, false);
    { const float4* p0 = (const float4*)(Wv + wrow + k0);
      const float4* p1 = (const float4*)(Wv + wrow + k1);
      a.h[0] = cvt8(p0[0], p0[1]); a.h[1] = cvt8(p1[0], p1[1]); }
    av = __builtin_amdgcn_wmma_f32_16x16x32_f16(false, a.v, false, bx.v, (short)0, av, false, false);
  }

  // C/D layout: VGPR r holds M = r + hi*8 (out channel offset), N = l = row16.
  const int l = l0 + row16;
  v8h qs, ks2;
#pragma unroll
  for (int r = 0; r < 8; ++r) {
    const int o = ow + hi * 8 + r;
    qs[r]  = (_Float16)(aq[r] + bq[o]);
    ks2[r] = (_Float16)(ak[r] + bk[o]);
  }
  // q,k: [b][l][c] — the 8 values per lane are contiguous in c.
  *(v8h*)(qh + ((size_t)b * LLEN + l) * CCH + ow + hi * 8) = qs;
  *(v8h*)(kh + ((size_t)b * LLEN + l) * CCH + ow + hi * 8) = ks2;
  // v: [b][c][l] (strided per r).
#pragma unroll
  for (int r = 0; r < 8; ++r) {
    const int o = ow + hi * 8 + r;
    vh[((size_t)b * CCH + o) * LLEN + l] = (_Float16)(av[r] + bv[o]);
  }
}

// ---------------------------------------------------------------------------
// Kernel 2: flash attention. Grid: B*(L/64) blocks, 128 threads (4 waves).
// Wave w owns queries [i0+16w, i0+16w+16). Loop over 64 key blocks of 32.
// K/V tiles double-buffered in LDS, filled by TDM (tensor_load_to_lds).
// ---------------------------------------------------------------------------
__global__ __launch_bounds__(128) void attn_kernel(
    const _Float16* __restrict__ qh, const _Float16* __restrict__ kh,
    const _Float16* __restrict__ vh, const float* __restrict__ x,
    const float* __restrict__ gamma, float* __restrict__ out)
{
  __shared__ _Float16 ks[2][32][136];   // K tile: [j][c], TDM-padded rows
  __shared__ _Float16 vs[2][128][40];   // V tile: [c][j], TDM-padded rows
  __shared__ _Float16 pw[4][16][32];    // per-wave P relayout scratch

  const int blk   = blockIdx.x;
  const int b     = blk / (LLEN / 64);
  const int i0blk = (blk % (LLEN / 64)) * 64;
  const int t     = threadIdx.x;
  const int wave  = t >> 5;
  const int lane  = t & 31;
  const int row16 = lane & 15;
  const int hi    = lane >> 4;
  const int i0    = i0blk + wave * 16;

  const _Float16* kbase = kh + (size_t)b * LLEN * CCH;   // [l][c]
  const _Float16* vbase = vh + (size_t)b * CCH * LLEN;   // [c][l]

  // Preload Q A-fragments (16 queries x 128 channels = 4 fragments of K=32).
  V16 qf[4];
  const _Float16* qrow = qh + ((size_t)b * LLEN + i0 + row16) * CCH;
#pragma unroll
  for (int ccb = 0; ccb < 4; ++ccb) {
    qf[ccb].h[0] = *(const v8h*)(qrow + ccb * 32 + hi * 8);
    qf[ccb].h[1] = *(const v8h*)(qrow + ccb * 32 + 16 + hi * 8);
  }

  v8f o[8];
#pragma unroll
  for (int i = 0; i < 8; ++i) o[i] = (v8f){};
  float mrun[8], lrun[8];
#pragma unroll
  for (int r = 0; r < 8; ++r) { mrun[r] = -1e30f; lrun[r] = 0.0f; }

  // TDM tile issue: K tile 32x128 (stride 128, pad 8 halfs / 128 halfs -> iv=5,am=3),
  //                 V tile 128x32 (stride 2048, pad 8 halfs / 32 halfs -> iv=3,am=3).
  auto issue_tiles = [&](int jb, int buf) {
    tdm_load_2d(kbase + (size_t)jb * 32 * CCH, (unsigned)(uintptr_t)&ks[buf][0][0],
                128u, 32u, (unsigned)CCH, 5u, 3u);
    tdm_load_2d(vbase + jb * 32, (unsigned)(uintptr_t)&vs[buf][0][0],
                32u, 128u, (unsigned)LLEN, 3u, 3u);
  };

  if (wave == 0) {
    issue_tiles(0, 0);
    __builtin_amdgcn_s_wait_tensorcnt(0);
  }
  __syncthreads();

  for (int jb = 0; jb < LLEN / 32; ++jb) {
    const int cur = jb & 1;
    // Prefetch next j-block into the other buffer while we compute on this one.
    if (wave == 0 && jb + 1 < LLEN / 32) issue_tiles(jb + 1, cur ^ 1);

    // S = q^T k : two 16x16 tiles (key halves), K-reduction over C=128.
    v8f S0 = {}, S1 = {};
#pragma unroll
    for (int ccb = 0; ccb < 4; ++ccb) {
      V16 kb0, kb1;
      kb0.h[0] = *(const v8h*)&ks[cur][row16][ccb * 32 + hi * 8];
      kb0.h[1] = *(const v8h*)&ks[cur][row16][ccb * 32 + 16 + hi * 8];
      kb1.h[0] = *(const v8h*)&ks[cur][16 + row16][ccb * 32 + hi * 8];
      kb1.h[1] = *(const v8h*)&ks[cur][16 + row16][ccb * 32 + 16 + hi * 8];
      S0 = __builtin_amdgcn_wmma_f32_16x16x32_f16(false, qf[ccb].v, false, kb0.v, (short)0, S0, false, false);
      S1 = __builtin_amdgcn_wmma_f32_16x16x32_f16(false, qf[ccb].v, false, kb1.v, (short)0, S1, false, false);
    }

    // Online softmax. Row i = r + hi*8 lives in VGPR r across 16 lanes;
    // xor-masks 1,2,4,8 stay inside each 16-lane half.
    float mnew[8];
#pragma unroll
    for (int r = 0; r < 8; ++r) mnew[r] = fmaxf(S0[r], S1[r]);
#pragma unroll
    for (int m = 1; m < 16; m <<= 1)
#pragma unroll
      for (int r = 0; r < 8; ++r) mnew[r] = fmaxf(mnew[r], __shfl_xor(mnew[r], m, 32));

    float alpha[8];
#pragma unroll
    for (int r = 0; r < 8; ++r) {
      const float mf = fmaxf(mrun[r], mnew[r]);
      alpha[r] = __expf(mrun[r] - mf);
      mrun[r]  = mf;
    }
    float p0[8], p1[8], rs[8];
#pragma unroll
    for (int r = 0; r < 8; ++r) {
      p0[r] = __expf(S0[r] - mrun[r]);
      p1[r] = __expf(S1[r] - mrun[r]);
      rs[r] = p0[r] + p1[r];
    }
#pragma unroll
    for (int m = 1; m < 16; m <<= 1)
#pragma unroll
      for (int r = 0; r < 8; ++r) rs[r] += __shfl_xor(rs[r], m, 32);
#pragma unroll
    for (int r = 0; r < 8; ++r) lrun[r] = lrun[r] * alpha[r] + rs[r];
#pragma unroll
    for (int c8 = 0; c8 < 8; ++c8)
#pragma unroll
      for (int r = 0; r < 8; ++r) o[c8][r] *= alpha[r];

    // Relayout P (C-layout in regs) -> A-fragment via per-wave LDS scratch.
#pragma unroll
    for (int r = 0; r < 8; ++r) {
      pw[wave][r + hi * 8][row16]      = (_Float16)p0[r];
      pw[wave][r + hi * 8][16 + row16] = (_Float16)p1[r];
    }
    asm volatile("" ::: "memory");   // same-wave DS ops are in-order; just stop reordering
    V16 pa;
    pa.h[0] = *(const v8h*)&pw[wave][row16][hi * 8];
    pa.h[1] = *(const v8h*)&pw[wave][row16][16 + hi * 8];

    // O(16 x 128) += P(16x32) * V^T(32x128), 8 N-chunks of 16 channels.
#pragma unroll
    for (int c8 = 0; c8 < 8; ++c8) {
      V16 vb;
      vb.h[0] = *(const v8h*)&vs[cur][c8 * 16 + row16][hi * 8];
      vb.h[1] = *(const v8h*)&vs[cur][c8 * 16 + row16][16 + hi * 8];
      o[c8] = __builtin_amdgcn_wmma_f32_16x16x32_f16(false, pa.v, false, vb.v, (short)0, o[c8], false, false);
    }

    // Wait for the prefetched tiles, then release both buffers to all waves.
    if (wave == 0) __builtin_amdgcn_s_wait_tensorcnt(0);
    __syncthreads();
  }

  // Finalize: divide by softmax denominator, apply gamma, add residual x.
  const float g = gamma[0];
  float inv[8];
#pragma unroll
  for (int r = 0; r < 8; ++r) inv[r] = 1.0f / lrun[r];
#pragma unroll
  for (int c8 = 0; c8 < 8; ++c8) {
#pragma unroll
    for (int r = 0; r < 8; ++r) {
      const int c = c8 * 16 + row16;
      const int i = i0 + r + hi * 8;
      const size_t idx = ((size_t)b * CCH + c) * LLEN + i;
      out[idx] = g * (o[c8][r] * inv[r]) + x[idx];
    }
  }
}

// ---------------------------------------------------------------------------
extern "C" void kernel_launch(void* const* d_in, const int* in_sizes, int n_in,
                              void* d_out, int out_size, void* d_ws, size_t ws_size,
                              hipStream_t stream) {
  (void)in_sizes; (void)n_in; (void)out_size; (void)ws_size;
  const float* x     = (const float*)d_in[0];
  const float* Wq    = (const float*)d_in[1];
  const float* bq    = (const float*)d_in[2];
  const float* Wk    = (const float*)d_in[3];
  const float* bk    = (const float*)d_in[4];
  const float* Wv    = (const float*)d_in[5];
  const float* bv    = (const float*)d_in[6];
  const float* gamma = (const float*)d_in[7];
  float* out = (float*)d_out;

  // Workspace: q, k, v in f16 (16 MB each, 48 MB total).
  _Float16* qh = (_Float16*)d_ws;
  _Float16* kh = qh + (size_t)BB * LLEN * CCH;
  _Float16* vh = kh + (size_t)BB * LLEN * CCH;

  qkv_proj_kernel<<<BB * (LLEN / 16), 256, 0, stream>>>(x, Wq, bq, Wk, bk, Wv, bv, qh, kh, vh);
  attn_kernel<<<BB * (LLEN / 64), 128, 0, stream>>>(qh, kh, vh, x, gamma, out);
}
